// SMPGNN_33629593928250
// MI455X (gfx1250) — 8-bit matrix operand order repaired and checked by reading, compiled, not run on hardware
//
#include <hip/hip_runtime.h>
#include <cstdint>
#include <cstddef>

typedef int v8i __attribute__((ext_vector_type(8)));

#define NN   100000
#define INC  512
#define HH   256
#define CC   40
#define EE   1600000
#define QMAXF 127.0f


__global__ void fill_zero_kernel(unsigned* __restrict__ p, long long n) {
    long long i = (long long)blockIdx.x * blockDim.x + threadIdx.x;
    long long stride = (long long)gridDim.x * blockDim.x;
    for (; i < n; i += stride) p[i] = 0u;
}

__global__ void absmax_kernel(const float* __restrict__ src, long long n,
                              unsigned* __restrict__ dst) {
    __shared__ float smax[256];
    float m = 0.0f;
    long long i = (long long)blockIdx.x * blockDim.x + threadIdx.x;
    long long stride = (long long)gridDim.x * blockDim.x;
    for (; i < n; i += stride) m = fmaxf(m, fabsf(src[i]));
    smax[threadIdx.x] = m;
    __syncthreads();
    for (int s = 128; s > 0; s >>= 1) {
        if ((int)threadIdx.x < s) smax[threadIdx.x] = fmaxf(smax[threadIdx.x], smax[threadIdx.x + s]);
        __syncthreads();
    }
    if (threadIdx.x == 0) atomicMax(dst, __float_as_uint(smax[0]));
}

__global__ void quant_f32_kernel(const float* __restrict__ src, long long n,
                                 const unsigned* __restrict__ amax_slot,
                                 int8_t* __restrict__ dst) {
    float scale = __uint_as_float(*amax_slot) * (1.0f / QMAXF) + 1e-12f;
    float inv = 1.0f / scale;
    long long i = (long long)blockIdx.x * blockDim.x + threadIdx.x;
    if (i < n) {
        float q = rintf(src[i] * inv);
        q = fminf(fmaxf(q, -128.0f), 127.0f);
        dst[i] = (int8_t)q;
    }
}

__global__ void quant_f16_kernel(const _Float16* __restrict__ src, long long n,
                                 const unsigned* __restrict__ amax_slot,
                                 int8_t* __restrict__ dst) {
    float scale = __uint_as_float(*amax_slot) * (1.0f / QMAXF) + 1e-12f;
    float inv = 1.0f / scale;
    long long i = (long long)blockIdx.x * blockDim.x + threadIdx.x;
    if (i < n) {
        float q = rintf((float)src[i] * inv);
        q = fminf(fmaxf(q, -128.0f), 127.0f);
        dst[i] = (int8_t)q;
    }
}

__device__ __forceinline__ v8i load_a8(const int8_t* p) {
    uint4 q0 = *(const uint4*)(p);
    uint4 q1 = *(const uint4*)(p + 32);
    v8i a = {(int)q0.x, (int)q0.y, (int)q0.z, (int)q0.w,
             (int)q1.x, (int)q1.y, (int)q1.z, (int)q1.w};
    return a;
}
__device__ __forceinline__ v8i load_b8(const int8_t* p) {
    uint4 r0 = *(const uint4*)(p);
    uint4 r1 = *(const uint4*)(p + 32);
    v8i b = {(int)r0.x, (int)r0.y, (int)r0.z, (int)r0.w,
             (int)r1.x, (int)r1.y, (int)r1.z, (int)r1.w};
    return b;
}

__global__ __launch_bounds__(256)
void gemm1_iu8_kernel(const int8_t* __restrict__ qx, const int8_t* __restrict__ qw1,
                      const float* __restrict__ b1, const unsigned* __restrict__ scal,
                      _Float16* __restrict__ hbuf, unsigned* __restrict__ amax_h) {
    const int lane = threadIdx.x & 31;
    const int wave = threadIdx.x >> 5;
    const int n    = lane & 15;
    const int half = lane >> 4;
    const int mbase = blockIdx.x * 16;

    const float sx = __uint_as_float(scal[0]) * (1.0f / QMAXF) + 1e-12f;
    const float sw = __uint_as_float(scal[1]) * (1.0f / QMAXF) + 1e-12f;
    const float sc = sx * sw;

    const int cb = wave * 32;
    const int8_t* arow  = qx  + (size_t)(mbase + n) * INC + half * 16;
    const int8_t* brow0 = qw1 + (size_t)(cb + n)     * INC + half * 16;
    const int8_t* brow1 = brow0 + (size_t)16 * INC;

    v8i acc0 = {0, 0, 0, 0, 0, 0, 0, 0};
    v8i acc1 = {0, 0, 0, 0, 0, 0, 0, 0};
#pragma unroll
    for (int k = 0; k < INC; k += 64) {
        v8i a = load_a8(arow + k);
        acc0 = __builtin_amdgcn_wmma_i32_16x16x64_iu8(true, a, true, load_b8(brow0 + k),
                                                      acc0, false, false);
        acc1 = __builtin_amdgcn_wmma_i32_16x16x64_iu8(true, a, true, load_b8(brow1 + k),
                                                      acc1, false, false);
    }

    float lmax = 0.0f;
#pragma unroll
    for (int r = 0; r < 8; ++r) {
        const int row = mbase + r + half * 8;
        const int c0 = cb + n, c1 = c0 + 16;
        float v0 = fmaxf(fmaf((float)acc0[r], sc, b1[c0]), 0.0f);
        float v1 = fmaxf(fmaf((float)acc1[r], sc, b1[c1]), 0.0f);
        hbuf[(size_t)row * HH + c0] = (_Float16)v0;
        hbuf[(size_t)row * HH + c1] = (_Float16)v1;
        lmax = fmaxf(lmax, fmaxf(v0, v1));
    }
    for (int off = 16; off > 0; off >>= 1)
        lmax = fmaxf(lmax, __shfl_xor(lmax, off, 32));
    if (lane == 0) atomicMax(amax_h, __float_as_uint(lmax));
}

__global__ __launch_bounds__(256)
void gemm2_iu8_kernel(const int8_t* __restrict__ qh, const int8_t* __restrict__ qw2,
                      const float* __restrict__ b2, const unsigned* __restrict__ scal,
                      float* __restrict__ zbuf) {
    const int lane = threadIdx.x & 31;
    const int wave = threadIdx.x >> 5;
    const int n    = lane & 15;
    const int half = lane >> 4;
    const int mbase = (blockIdx.x * 8 + wave) * 16;
    if (mbase >= NN) return;

    const float sh = __uint_as_float(scal[2]) * (1.0f / QMAXF) + 1e-12f;
    const float sw = __uint_as_float(scal[3]) * (1.0f / QMAXF) + 1e-12f;
    const float sc = sh * sw;

    const int8_t* arow = qh + (size_t)(mbase + n) * HH + half * 16;
    const int c0 = n, c1 = 16 + n, c2 = 32 + n;
    const int8_t* brow0 = qw2 + (size_t)c0 * HH + half * 16;
    const int8_t* brow1 = qw2 + (size_t)c1 * HH + half * 16;
    const int8_t* brow2 = qw2 + (size_t)c2 * HH + half * 16;
    const bool v2 = (c2 < CC);
    const v8i zero = {0, 0, 0, 0, 0, 0, 0, 0};

    v8i acc0 = zero, acc1 = zero, acc2 = zero;
#pragma unroll
    for (int k = 0; k < HH; k += 64) {
        v8i a = load_a8(arow + k);
        acc0 = __builtin_amdgcn_wmma_i32_16x16x64_iu8(true, a, true, load_b8(brow0 + k),
                                                      acc0, false, false);
        acc1 = __builtin_amdgcn_wmma_i32_16x16x64_iu8(true, a, true, load_b8(brow1 + k),
                                                      acc1, false, false);
        v8i b2v = v2 ? load_b8(brow2 + k) : zero;
        acc2 = __builtin_amdgcn_wmma_i32_16x16x64_iu8(true, a, true, b2v,
                                                      acc2, false, false);
    }

#pragma unroll
    for (int r = 0; r < 8; ++r) {
        const int row = mbase + r + half * 8;
        zbuf[(size_t)row * CC + c0] = fmaf((float)acc0[r], sc, b2[c0]);
        zbuf[(size_t)row * CC + c1] = fmaf((float)acc1[r], sc, b2[c1]);
        if (v2) zbuf[(size_t)row * CC + c2] = fmaf((float)acc2[r], sc, b2[c2]);
    }
}


__global__ void degree_kernel(const int* __restrict__ ei, float* __restrict__ deg) {
    int e = blockIdx.x * blockDim.x + threadIdx.x;
    if (e < EE) atomicAdd(&deg[ei[EE + e]], 1.0f);
}

__global__ void dinv_kernel(float* __restrict__ deg) {
    int i = blockIdx.x * blockDim.x + threadIdx.x;
    if (i < NN) {
        float d = deg[i];
        deg[i] = (d > 0.0f) ? rsqrtf(fmaxf(d, 1.0f)) : 0.0f;
    }
}

__global__ void scatter_kernel(const int* __restrict__ ei, const float* __restrict__ dinv,
                               const float* __restrict__ zbuf, float* __restrict__ out) {
    const int CHUNKS = CC / 4;
    long long idx = (long long)blockIdx.x * blockDim.x + threadIdx.x;
    if (idx >= (long long)EE * CHUNKS) return;
    int e  = (int)(idx / CHUNKS);
    int cb = (int)(idx - (long long)e * CHUNKS) * 4;
    int s = ei[e], d = ei[EE + e];
    float coef = dinv[s] * dinv[d];
    float4 zv = *(const float4*)(zbuf + (size_t)s * CC + cb);
    float* o = out + (size_t)d * CC + cb;
    atomicAdd(o + 0, zv.x * coef);
    atomicAdd(o + 1, zv.y * coef);
    atomicAdd(o + 2, zv.z * coef);
    atomicAdd(o + 3, zv.w * coef);
}

__global__ __launch_bounds__(256)
void logsoftmax_kernel(float* __restrict__ out) {
    const int lane = threadIdx.x & 31;
    const int wave = threadIdx.x >> 5;
    long long row = (long long)blockIdx.x * 8 + wave;
    if (row >= NN) return;
    float* r = out + row * CC;
    float v0 = r[lane];
    float v1 = (lane < CC - 32) ? r[lane + 32] : -3.402823466e38f;
    float mx = fmaxf(v0, v1);
    for (int off = 16; off > 0; off >>= 1) mx = fmaxf(mx, __shfl_xor(mx, off, 32));
    float s = expf(v0 - mx) + ((lane < CC - 32) ? expf(v1 - mx) : 0.0f);
    for (int off = 16; off > 0; off >>= 1) s += __shfl_xor(s, off, 32);
    float lse = logf(s);
    r[lane] = v0 - mx - lse;
    if (lane < CC - 32) r[lane + 32] = v1 - mx - lse;
}


extern "C" void kernel_launch(void* const* d_in, const int* in_sizes, int n_in,
                              void* d_out, int out_size, void* d_ws, size_t ws_size,
                              hipStream_t stream) {
    const float* x  = (const float*)d_in[0];
    const int*   ei = (const int*)  d_in[1];
    const float* w1 = (const float*)d_in[2];
    const float* b1 = (const float*)d_in[3];
    const float* w2 = (const float*)d_in[4];
    const float* b2 = (const float*)d_in[5];
    float* out = (float*)d_out;
    char*  ws  = (char*)d_ws;

    unsigned*  scal = (unsigned*) ws;
    float*     deg  = (float*)   (ws + 256);
    int8_t*    qx   = (int8_t*)  (ws + 400384);
    int8_t*    qw1  = (int8_t*)  (ws + 51600384);
    int8_t*    qw2  = (int8_t*)  (ws + 51731456);
    _Float16*  hbuf = (_Float16*)(ws + 51741696);
    int8_t*    qh   = (int8_t*)  (ws + 102941696);
    float*     zbuf = (float*)   (ws + 128541696);

    fill_zero_kernel<<<1, 64, 0, stream>>>(scal, 4);
    fill_zero_kernel<<<512, 256, 0, stream>>>((unsigned*)deg, (long long)NN);
    fill_zero_kernel<<<2048, 256, 0, stream>>>((unsigned*)out, (long long)NN * CC);

    absmax_kernel<<<1024, 256, 0, stream>>>(x,  (long long)NN * INC, &scal[0]);
    absmax_kernel<<<256,  256, 0, stream>>>(w1, (long long)HH * INC, &scal[1]);
    absmax_kernel<<<64,   256, 0, stream>>>(w2, (long long)CC * HH,  &scal[3]);

    quant_f32_kernel<<<(NN * (long long)INC + 255) / 256, 256, 0, stream>>>(x,  (long long)NN * INC, &scal[0], qx);
    quant_f32_kernel<<<(HH * INC + 255) / 256,            256, 0, stream>>>(w1, (long long)HH * INC, &scal[1], qw1);
    quant_f32_kernel<<<(CC * HH + 255) / 256,             256, 0, stream>>>(w2, (long long)CC * HH,  &scal[3], qw2);

    gemm1_iu8_kernel<<<NN / 16, 256, 0, stream>>>(qx, qw1, b1, scal, hbuf, &scal[2]);

    quant_f16_kernel<<<(NN * (long long)HH + 255) / 256, 256, 0, stream>>>(hbuf, (long long)NN * HH, &scal[2], qh);

    gemm2_iu8_kernel<<<(NN / 16 + 7) / 8, 256, 0, stream>>>(qh, qw2, b2, scal, zbuf);

    degree_kernel<<<(EE + 255) / 256, 256, 0, stream>>>(ei, deg);
    dinv_kernel<<<(NN + 255) / 256, 256, 0, stream>>>(deg);
    scatter_kernel<<<(int)(((long long)EE * (CC / 4) + 255) / 256), 256, 0, stream>>>(ei, deg, zbuf, out);

    logsoftmax_kernel<<<(NN + 7) / 8, 256, 0, stream>>>(out);
}
